// AutoEncoder_42460046688392
// MI455X (gfx1250) — matrix operand order repaired and checked by reading, not compiled, not run on hardware
//
#include <hip/hip_runtime.h>
#include <hip/hip_bf16.h>

// ===========================================================================
// VQ-VAE forward for MI455X (gfx1250, wave32, WMMA).
//
// All conv layers are implicit-GEMM on v_wmma_f32_16x16x32_bf16 with 2D
// register blocking: each wave computes 32 pixels x 64 couts = 2x4 WMMA tiles
// (8 fp32 accumulators). Each B (weight) fragment feeds 2 WMMAs and each A
// (im2col) fragment feeds 4, i.e. 1.5 b128 loads per WMMA instead of 2.5 --
// this halves the dominant L2 weight-fragment traffic (~2.4 GB -> 1.2 GB per
// 3x3 conv layer), which is the real limiter since all tensors are resident
// in the 192 MB L2.
// Transposed convs are decomposed into 4 output-parity sub-convolutions.
// VQ argmin runs as a WMMA GEMM (16 pixels x 64 codes per wave) + in-register
// min across tiles + shfl_xor min-reduce + one u64 atomicMin per pixel/wave.
//
// Roofline: ~320 GFLOP of GEMM vs ~1-2 GB HBM traffic -> bf16 WMMA keeps
// compute (~100us at multi-PFLOPS bf16 dense) balanced against 23.3 TB/s HBM
// (~60-90us).
// ===========================================================================

typedef __attribute__((ext_vector_type(16))) __bf16 v16bf;
typedef __attribute__((ext_vector_type(8)))  float  v8f;

struct U32x8 { uint4 a; uint4 b; };

static __device__ __forceinline__ v16bf as_v16bf(uint4 a, uint4 b) {
  U32x8 u; u.a = a; u.b = b;
  return __builtin_bit_cast(v16bf, u);
}

static __device__ __forceinline__ unsigned short f2bf(float f) {
  unsigned u = __builtin_bit_cast(unsigned, f);
  unsigned r = u + 0x7FFFu + ((u >> 16) & 1u);   // round-to-nearest-even
  return (unsigned short)(r >> 16);
}

static __device__ __forceinline__ float bf2f(unsigned short b) {
  return __builtin_bit_cast(float, (unsigned)b << 16);
}

// ReLU on two packed bf16 values in one dword (clear negatives)
static __device__ __forceinline__ unsigned bfrelu2(unsigned x) {
  unsigned lo = x & 0x0000FFFFu;
  unsigned hi = x & 0xFFFF0000u;
  if (lo & 0x00008000u) lo = 0u;
  if (hi & 0x80000000u) hi = 0u;
  return lo | hi;
}
static __device__ __forceinline__ uint4 relu4(uint4 a) {
  a.x = bfrelu2(a.x); a.y = bfrelu2(a.y); a.z = bfrelu2(a.z); a.w = bfrelu2(a.w);
  return a;
}

// ---------------------------------------------------------------------------
// Generic fused conv-GEMM kernel parameters
// ---------------------------------------------------------------------------
struct ConvParams {
  const unsigned short* in;    // NHWC bf16 [N,H,W,Cin]
  const unsigned short* wt;    // packed bf16 [CoutPad][T*Cin]
  const float* bias;           // [CoutReal] or null
  const unsigned short* res;   // residual NHWC bf16 [N,OH,OW,CoutReal] or null
  unsigned short* out;         // NHWC bf16 output (full grid geometry) or null
  float* out2;                 // fp32 NCHW secondary output or null
  int N, H, W, Cin;
  int OH, OW, CoutReal;        // OH/OW: virtual (parity) output grid
  int stride, T, in_relu, act; // act: 0=none 1=relu 2=sigmoid
  int dy[16], dx[16];          // tap offsets: ih = oh*stride + dy[t]
  int oFH, oFW, ors, oro, ocs, oco;          // bf16 out geometry (full grid)
  int o2C, o2H, o2W, o2rs, o2ro, o2cs, o2co; // fp32 NCHW out geometry
};

// MT = 16-pixel tiles per wave, NT = 16-cout tiles per wave
template <int MT, int NT>
__global__ __launch_bounds__(256)
void conv_wmma_kernel(ConvParams P, int totalWaves, int coutGroups) {
  int wave = blockIdx.x * 8 + (threadIdx.x >> 5);
  if (wave >= totalWaves) return;                  // wave-uniform exit
  int pixGroup = wave / coutGroups;
  int cg       = wave - pixGroup * coutGroups;
  int lane = threadIdx.x & 31;
  int l16  = lane & 15;
  int h    = lane >> 4;

  int ohw = P.OH * P.OW;
  // A-fragment rows: pixel tile m, row l16 -> pixel (pixGroup*MT+m)*16 + l16
  int nidx[MT], ohm[MT], owm[MT];
#pragma unroll
  for (int m = 0; m < MT; ++m) {
    int pa  = (pixGroup * MT + m) * 16 + l16;
    int nn  = pa / ohw;
    int rem = pa - nn * ohw;
    int o_h = rem / P.OW;
    nidx[m] = nn; ohm[m] = o_h; owm[m] = rem - o_h * P.OW;
  }

  size_t wstride = (size_t)(P.T * P.Cin);
  const unsigned short* wrow[NT];
#pragma unroll
  for (int nt = 0; nt < NT; ++nt)
    wrow[nt] = P.wt + (size_t)(cg * (16 * NT) + nt * 16 + l16) * wstride;

  v8f acc[MT][NT];
#pragma unroll
  for (int m = 0; m < MT; ++m)
#pragma unroll
    for (int nt = 0; nt < NT; ++nt)
      acc[m][nt] = (v8f){0.f, 0.f, 0.f, 0.f, 0.f, 0.f, 0.f, 0.f};

  const uint4 z4 = make_uint4(0u, 0u, 0u, 0u);

  for (int t = 0; t < P.T; ++t) {                  // taps (uniform)
    const unsigned short* arow[MT];
    bool inb[MT];
#pragma unroll
    for (int m = 0; m < MT; ++m) {
      int ih = ohm[m] * P.stride + P.dy[t];
      int iw = owm[m] * P.stride + P.dx[t];
      inb[m] = ((unsigned)ih < (unsigned)P.H) && ((unsigned)iw < (unsigned)P.W);
      arow[m] = P.in +
          (((size_t)nidx[m] * P.H + (inb[m] ? ih : 0)) * P.W + (inb[m] ? iw : 0)) * P.Cin;
    }
#pragma unroll 2
    for (int c0 = 0; c0 < P.Cin; c0 += 32) {       // K chunks of 32
      // ---- issue all loads of this chunk ----
      uint4 a0[MT], a1[MT];
#pragma unroll
      for (int m = 0; m < MT; ++m) {
        a0[m] = z4; a1[m] = z4;
        if (inb[m]) {
          // A lane holds K = 16h..16h+15 (contiguous, matching B)
          a0[m] = *(const uint4*)(arow[m] + c0 + 16 * h);
          a1[m] = *(const uint4*)(arow[m] + c0 + 16 * h + 8);
        }
      }
      uint4 b0[NT], b1[NT];
      int kk = t * P.Cin + c0 + 16 * h;
#pragma unroll
      for (int nt = 0; nt < NT; ++nt) {
        // B lane holds K = 16h..16h+15 of its column (contiguous 32B)
        b0[nt] = *(const uint4*)(wrow[nt] + kk);
        b1[nt] = *(const uint4*)(wrow[nt] + kk + 8);
      }
      v16bf av[MT];
#pragma unroll
      for (int m = 0; m < MT; ++m) {
        uint4 ua0 = a0[m], ua1 = a1[m];
        if (P.in_relu) { ua0 = relu4(ua0); ua1 = relu4(ua1); }
        av[m] = as_v16bf(ua0, ua1);
      }
      // ---- consume: each B fragment feeds MT WMMAs ----
#pragma unroll
      for (int nt = 0; nt < NT; ++nt) {
        v16bf bv = as_v16bf(b0[nt], b1[nt]);
#pragma unroll
        for (int m = 0; m < MT; ++m)
          acc[m][nt] = __builtin_amdgcn_wmma_f32_16x16x32_bf16(
              false, av[m], false, bv, (short)0, acc[m][nt], false, false);
      }
    }
  }

  // Epilogue: C VGPR j -> (pixel = tile*16 + j + 8h, cout tile col = l16)
#pragma unroll
  for (int m = 0; m < MT; ++m) {
#pragma unroll
    for (int nt = 0; nt < NT; ++nt) {
      int co  = cg * (16 * NT) + nt * 16 + l16;
      if (co >= P.CoutReal) continue;
      float bv = P.bias ? P.bias[co] : 0.0f;
#pragma unroll
      for (int j = 0; j < 8; ++j) {
        int pe = (pixGroup * MT + m) * 16 + j + 8 * h;
        int n2 = pe / ohw;
        int r2 = pe - n2 * ohw;
        int oh2 = r2 / P.OW;
        int ow2 = r2 - oh2 * P.OW;
        float v = acc[m][nt][j] + bv;
        if (P.res) v += bf2f(P.res[(size_t)pe * P.CoutReal + co]);
        if (P.act == 1)      v = fmaxf(v, 0.0f);
        else if (P.act == 2) v = 1.0f / (1.0f + __expf(-v));
        if (P.out) {
          int ohf = oh2 * P.ors + P.oro;
          int owf = ow2 * P.ocs + P.oco;
          P.out[(((size_t)n2 * P.oFH + ohf) * P.oFW + owf) * P.CoutReal + co] = f2bf(v);
        }
        if (P.out2) {
          int oh2f = oh2 * P.o2rs + P.o2ro;
          int ow2f = ow2 * P.o2cs + P.o2co;
          P.out2[(((size_t)n2 * P.o2C + co) * P.o2H + oh2f) * P.o2W + ow2f] = v;
        }
      }
    }
  }
}

// ---------------------------------------------------------------------------
// VQ: argmin_k ( e2[k] - 2 * z . e_k )  via WMMA 16x(4x16) tiles per wave,
// in-register min across the 4 code tiles, shfl_xor min-reduce over 16 lanes,
// then one u64 atomicMin per pixel per wave.
// ---------------------------------------------------------------------------
__global__ __launch_bounds__(256)
void vq_argmin_kernel(const unsigned short* z, const unsigned short* embb,
                      const float* e2, unsigned long long* best,
                      int totalWaves, int codeGroups) {
  int wave = blockIdx.x * 8 + (threadIdx.x >> 5);
  if (wave >= totalWaves) return;
  int pixTile = wave / codeGroups;
  int cg      = wave - pixTile * codeGroups;
  int lane = threadIdx.x & 31;
  int l16  = lane & 15;
  int h    = lane >> 4;

  const unsigned short* zrow = z + (size_t)(pixTile * 16 + l16) * 256;
  const unsigned short* erow[4];
#pragma unroll
  for (int nt = 0; nt < 4; ++nt)
    erow[nt] = embb + (size_t)(cg * 64 + nt * 16 + l16) * 256;

  v8f acc[4];
#pragma unroll
  for (int nt = 0; nt < 4; ++nt)
    acc[nt] = (v8f){0.f, 0.f, 0.f, 0.f, 0.f, 0.f, 0.f, 0.f};

#pragma unroll
  for (int c0 = 0; c0 < 256; c0 += 32) {
    uint4 a0 = *(const uint4*)(zrow + c0 + 16 * h);
    uint4 a1 = *(const uint4*)(zrow + c0 + 16 * h + 8);
    uint4 b0[4], b1[4];
#pragma unroll
    for (int nt = 0; nt < 4; ++nt) {
      b0[nt] = *(const uint4*)(erow[nt] + c0 + 16 * h);
      b1[nt] = *(const uint4*)(erow[nt] + c0 + 16 * h + 8);
    }
    v16bf av = as_v16bf(a0, a1);
#pragma unroll
    for (int nt = 0; nt < 4; ++nt)
      acc[nt] = __builtin_amdgcn_wmma_f32_16x16x32_bf16(
          false, av, false, as_v16bf(b0[nt], b1[nt]), (short)0,
          acc[nt], false, false);
  }

  float ek[4];
#pragma unroll
  for (int nt = 0; nt < 4; ++nt) ek[nt] = e2[cg * 64 + nt * 16 + l16];

#pragma unroll
  for (int j = 0; j < 8; ++j) {
    unsigned long long key = ~0ull;
#pragma unroll
    for (int nt = 0; nt < 4; ++nt) {
      int code = cg * 64 + nt * 16 + l16;
      float dist = ek[nt] - 2.0f * acc[nt][j];
      unsigned u = __builtin_bit_cast(unsigned, dist);
      unsigned k32 = (u & 0x80000000u) ? ~u : (u | 0x80000000u); // order-preserving
      unsigned long long k = ((unsigned long long)k32 << 32) | (unsigned)code;
      key = (k < key) ? k : key;
    }
    // min over the 16 lanes of this half-wave (16 columns each holding 4 codes)
#pragma unroll
    for (int m = 8; m >= 1; m >>= 1) {
      unsigned lo = __shfl_xor((unsigned)key, m, 32);
      unsigned hi = __shfl_xor((unsigned)(key >> 32), m, 32);
      unsigned long long o = ((unsigned long long)hi << 32) | lo;
      key = (o < key) ? o : key;
    }
    if (l16 == 0)
      atomicMin(best + (pixTile * 16 + j + 8 * h), key);
  }
}

__global__ __launch_bounds__(256)
void vq_gather_kernel(const unsigned long long* best, const float* emb,
                      unsigned short* zq, float* zqx) {
  int p = blockIdx.x;            // pixel (32*32*32 of them)
  int c = threadIdx.x;           // channel 0..255
  int idx = (int)(best[p] & 0xFFFFFFFFull);
  float v = emb[(size_t)idx * 256 + c];
  zq[(size_t)p * 256 + c] = f2bf(v);
  int n = p >> 10, r = p & 1023, hh = r >> 5, ww = r & 31;
  zqx[(((size_t)n * 256 + c) * 32 + hh) * 32 + ww] = v;
}

// ---------------------------------------------------------------------------
// Packing kernels
// ---------------------------------------------------------------------------
__global__ void pack_x_kernel(const float* x, unsigned short* xp, int total) {
  int i = blockIdx.x * 256 + threadIdx.x;
  if (i >= total) return;
  int c = i & 31, pix = i >> 5;
  int n = pix >> 14, r = pix & 16383, hh = r >> 7, ww = r & 127;
  float v = (c < 3) ? x[(((size_t)n * 3 + c) * 128 + hh) * 128 + ww] : 0.0f;
  xp[i] = f2bf(v);
}

// OIHW fp32 -> [co][kh*KW+kw][ci(padded)] bf16
__global__ void pack_w_kernel(const float* w, unsigned short* wt, int Cin,
                              int KH, int KW, int CinPad, int total) {
  int i = blockIdx.x * 256 + threadIdx.x;
  if (i >= total) return;
  int ci = i % CinPad;
  int r  = i / CinPad;
  int t  = r % (KH * KW);
  int co = r / (KH * KW);
  int kh = t / KW, kw = t % KW;
  float v = (ci < Cin) ? w[(((size_t)co * Cin + ci) * KH + kh) * KW + kw] : 0.0f;
  wt[i] = f2bf(v);
}

// ConvTranspose [Cin][Cout][4][4] fp32 -> [parity][co(padded)][tap][ci] bf16
// parity (po,pw); po==0: kh {1,3} <-> dy {0,-1}; po==1: kh {0,2} <-> dy {1,0}
__global__ void pack_tw_kernel(const float* w, unsigned short* wt, int Cin,
                               int Cout, int CoutPad, int total) {
  int i = blockIdx.x * 256 + threadIdx.x;
  if (i >= total) return;
  int ci = i % Cin;
  int r  = i / Cin;
  int t  = r & 3; r >>= 2;
  int co = r % CoutPad;
  int par = r / CoutPad;
  int po = par >> 1, pw = par & 1;
  int tdy = t >> 1, tdx = t & 1;
  int kh = po ? (tdy ? 2 : 0) : (tdy ? 3 : 1);
  int kw = pw ? (tdx ? 2 : 0) : (tdx ? 3 : 1);
  float v = (co < Cout) ? w[(((size_t)ci * Cout + co) * 4 + kh) * 4 + kw] : 0.0f;
  wt[i] = f2bf(v);
}

__global__ void pack_emb_kernel(const float* emb, unsigned short* embb, int total) {
  int i = blockIdx.x * 256 + threadIdx.x;
  if (i < total) embb[i] = f2bf(emb[i]);
}

__global__ void e2_kernel(const float* emb, float* e2) {
  int k = blockIdx.x * 64 + threadIdx.x;
  if (k >= 512) return;
  float s = 0.0f;
  for (int d = 0; d < 256; ++d) { float v = emb[(size_t)k * 256 + d]; s += v * v; }
  e2[k] = s;
}

// ---------------------------------------------------------------------------
// Host orchestration
// ---------------------------------------------------------------------------
static void run_conv(hipStream_t stream,
                     const unsigned short* in, int H, int W, int Cin,
                     const unsigned short* wt, const float* bias,
                     const unsigned short* res, unsigned short* out, float* out2,
                     int OH, int OW, int CoutReal, int CoutPad,
                     int stride, int T, const int* dys, const int* dxs,
                     int in_relu, int act,
                     int oFH, int oFW, int ors, int oro, int ocs, int oco,
                     int o2C, int o2H, int o2W, int o2rs, int o2ro, int o2cs, int o2co) {
  ConvParams P;
  P.in = in; P.wt = wt; P.bias = bias; P.res = res; P.out = out; P.out2 = out2;
  P.N = 32; P.H = H; P.W = W; P.Cin = Cin;
  P.OH = OH; P.OW = OW; P.CoutReal = CoutReal;
  P.stride = stride; P.T = T; P.in_relu = in_relu; P.act = act;
  for (int t = 0; t < 16; ++t) { P.dy[t] = (t < T) ? dys[t] : 0; P.dx[t] = (t < T) ? dxs[t] : 0; }
  P.oFH = oFH; P.oFW = oFW; P.ors = ors; P.oro = oro; P.ocs = ocs; P.oco = oco;
  P.o2C = o2C; P.o2H = o2H; P.o2W = o2W;
  P.o2rs = o2rs; P.o2ro = o2ro; P.o2cs = o2cs; P.o2co = o2co;
  int pixTiles  = (32 * OH * OW) / 16;
  int pixGroups = pixTiles / 2;   // MT = 2 (all layers have even pixTiles)
  if ((CoutPad & 63) == 0) {
    int coutGroups = CoutPad / 64;
    int totalWaves = pixGroups * coutGroups;
    int blocks     = (totalWaves + 7) / 8;
    conv_wmma_kernel<2, 4><<<blocks, 256, 0, stream>>>(P, totalWaves, coutGroups);
  } else {
    int coutGroups = CoutPad / 16;
    int totalWaves = pixGroups * coutGroups;
    int blocks     = (totalWaves + 7) / 8;
    conv_wmma_kernel<2, 1><<<blocks, 256, 0, stream>>>(P, totalWaves, coutGroups);
  }
}

extern "C" void kernel_launch(void* const* d_in, const int* in_sizes, int n_in,
                              void* d_out, int out_size, void* d_ws, size_t ws_size,
                              hipStream_t stream) {
  (void)in_sizes; (void)n_in; (void)out_size; (void)ws_size;

  const float* x     = (const float*)d_in[0];
  const float* ec1w  = (const float*)d_in[1];
  const float* ec1b  = (const float*)d_in[2];
  const float* ec2w  = (const float*)d_in[3];
  const float* ec2b  = (const float*)d_in[4];
  const float* er1w1 = (const float*)d_in[5];
  const float* er1b1 = (const float*)d_in[6];
  const float* er1w2 = (const float*)d_in[7];
  const float* er1b2 = (const float*)d_in[8];
  const float* er2w1 = (const float*)d_in[9];
  const float* er2b1 = (const float*)d_in[10];
  const float* er2w2 = (const float*)d_in[11];
  const float* er2b2 = (const float*)d_in[12];
  const float* emb   = (const float*)d_in[13];
  const float* dr1w1 = (const float*)d_in[14];
  const float* dr1b1 = (const float*)d_in[15];
  const float* dr1w2 = (const float*)d_in[16];
  const float* dr1b2 = (const float*)d_in[17];
  const float* dr2w1 = (const float*)d_in[18];
  const float* dr2b1 = (const float*)d_in[19];
  const float* dr2w2 = (const float*)d_in[20];
  const float* dr2b2 = (const float*)d_in[21];
  const float* dt1w  = (const float*)d_in[22];
  const float* dt1b  = (const float*)d_in[23];
  const float* dt2w  = (const float*)d_in[24];
  const float* dt2b  = (const float*)d_in[25];

  float* out_x  = (float*)d_out;                         // [32,3,128,128]
  float* out_ze = out_x + (size_t)32 * 3 * 128 * 128;    // [32,256,32,32]
  float* out_zq = out_ze + (size_t)32 * 256 * 32 * 32;   // [32,256,32,32]

  // ---- workspace bump allocator (~161 MB total) ----
  char* cur = (char*)d_ws;
  auto alloc = [&](size_t bytes) -> void* {
    void* p = (void*)cur;
    cur += (bytes + 255) & ~(size_t)255;
    return p;
  };
  unsigned short* xp   = (unsigned short*)alloc((size_t)32 * 128 * 128 * 32 * 2);
  unsigned short* bufA = (unsigned short*)alloc((size_t)32 * 64 * 64 * 256 * 2);
  unsigned short* b1   = (unsigned short*)alloc((size_t)32 * 32 * 32 * 256 * 2);
  unsigned short* b2   = (unsigned short*)alloc((size_t)32 * 32 * 32 * 256 * 2);
  unsigned short* b3   = (unsigned short*)alloc((size_t)32 * 32 * 32 * 256 * 2);
  unsigned short* wc1  = (unsigned short*)alloc((size_t)256 * 16 * 32 * 2);
  unsigned short* wc2  = (unsigned short*)alloc((size_t)256 * 16 * 256 * 2);
  unsigned short* wra[4]; // resblock 3x3 packed
  unsigned short* wrb[4]; // resblock 1x1 packed
  for (int i = 0; i < 4; ++i) {
    wra[i] = (unsigned short*)alloc((size_t)256 * 9 * 256 * 2);
    wrb[i] = (unsigned short*)alloc((size_t)256 * 1 * 256 * 2);
  }
  unsigned short* wt1  = (unsigned short*)alloc((size_t)4 * 256 * 4 * 256 * 2);
  unsigned short* wt2  = (unsigned short*)alloc((size_t)4 * 16 * 4 * 256 * 2);
  unsigned short* embb = (unsigned short*)alloc((size_t)512 * 256 * 2);
  float*          e2   = (float*)alloc((size_t)512 * 4);
  unsigned long long* best = (unsigned long long*)alloc((size_t)32768 * 8);

  // ---- packing ----
  { int total = 32 * 128 * 128 * 32;
    pack_x_kernel<<<(total + 255) / 256, 256, 0, stream>>>(x, xp, total); }
  auto packw = [&](const float* w, unsigned short* wtp, int Cin, int KH, int KW, int CinPad) {
    int total = 256 * KH * KW * CinPad;
    pack_w_kernel<<<(total + 255) / 256, 256, 0, stream>>>(w, wtp, Cin, KH, KW, CinPad, total);
  };
  packw(ec1w, wc1, 3, 4, 4, 32);
  packw(ec2w, wc2, 256, 4, 4, 256);
  packw(er1w1, wra[0], 256, 3, 3, 256);  packw(er1w2, wrb[0], 256, 1, 1, 256);
  packw(er2w1, wra[1], 256, 3, 3, 256);  packw(er2w2, wrb[1], 256, 1, 1, 256);
  packw(dr1w1, wra[2], 256, 3, 3, 256);  packw(dr1w2, wrb[2], 256, 1, 1, 256);
  packw(dr2w1, wra[3], 256, 3, 3, 256);  packw(dr2w2, wrb[3], 256, 1, 1, 256);
  { int total = 4 * 256 * 4 * 256;
    pack_tw_kernel<<<(total + 255) / 256, 256, 0, stream>>>(dt1w, wt1, 256, 256, 256, total); }
  { int total = 4 * 16 * 4 * 256;
    pack_tw_kernel<<<(total + 255) / 256, 256, 0, stream>>>(dt2w, wt2, 256, 3, 16, total); }
  { int total = 512 * 256;
    pack_emb_kernel<<<(total + 255) / 256, 256, 0, stream>>>(emb, embb, total); }
  e2_kernel<<<8, 64, 0, stream>>>(emb, e2);
  hipMemsetAsync(best, 0xFF, (size_t)32768 * 8, stream);

  // ---- tap tables ----
  int dy44[16], dx44[16];
  for (int kh = 0; kh < 4; ++kh)
    for (int kw = 0; kw < 4; ++kw) { dy44[kh * 4 + kw] = kh - 1; dx44[kh * 4 + kw] = kw - 1; }
  int dy33[9], dx33[9];
  for (int kh = 0; kh < 3; ++kh)
    for (int kw = 0; kw < 3; ++kw) { dy33[kh * 3 + kw] = kh - 1; dx33[kh * 3 + kw] = kw - 1; }
  int dy11[1] = {0}, dx11[1] = {0};

  // ---- encoder ----
  // conv1: 4x4 s2, Cin padded 3->32, relu -> bufA [32,64,64,256]
  run_conv(stream, xp, 128, 128, 32, wc1, ec1b, nullptr, bufA, nullptr,
           64, 64, 256, 256, 2, 16, dy44, dx44, 0, 1,
           64, 64, 1, 0, 1, 0, 0, 0, 0, 0, 0, 0, 0);
  // conv2: 4x4 s2, relu -> b1 [32,32,32,256]
  run_conv(stream, bufA, 64, 64, 256, wc2, ec2b, nullptr, b1, nullptr,
           32, 32, 256, 256, 2, 16, dy44, dx44, 0, 1,
           32, 32, 1, 0, 1, 0, 0, 0, 0, 0, 0, 0, 0);
  // enc resblock 1: b1 -> b3  (h=b2)
  run_conv(stream, b1, 32, 32, 256, wra[0], er1b1, nullptr, b2, nullptr,
           32, 32, 256, 256, 1, 9, dy33, dx33, 1, 1,
           32, 32, 1, 0, 1, 0, 0, 0, 0, 0, 0, 0, 0);
  run_conv(stream, b2, 32, 32, 256, wrb[0], er1b2, b1, b3, nullptr,
           32, 32, 256, 256, 1, 1, dy11, dx11, 0, 0,
           32, 32, 1, 0, 1, 0, 0, 0, 0, 0, 0, 0, 0);
  // enc resblock 2: b3 -> b2 (= z_e), also fp32 NCHW z_e_x into d_out
  run_conv(stream, b3, 32, 32, 256, wra[1], er2b1, nullptr, b1, nullptr,
           32, 32, 256, 256, 1, 9, dy33, dx33, 1, 1,
           32, 32, 1, 0, 1, 0, 0, 0, 0, 0, 0, 0, 0);
  run_conv(stream, b1, 32, 32, 256, wrb[1], er2b2, b3, b2, out_ze,
           32, 32, 256, 256, 1, 1, dy11, dx11, 0, 0,
           32, 32, 1, 0, 1, 0, 256, 32, 32, 1, 0, 1, 0);

  // ---- VQ ----
  { int totalWaves = (32768 / 16) * 8;   // pixTiles * codeGroups(512/64)
    vq_argmin_kernel<<<(totalWaves + 7) / 8, 256, 0, stream>>>(
        b2, embb, e2, best, totalWaves, 8); }
  vq_gather_kernel<<<32768, 256, 0, stream>>>(best, emb, b3, out_zq);

  // ---- decoder resblocks ----
  run_conv(stream, b3, 32, 32, 256, wra[2], dr1b1, nullptr, b1, nullptr,
           32, 32, 256, 256, 1, 9, dy33, dx33, 1, 1,
           32, 32, 1, 0, 1, 0, 0, 0, 0, 0, 0, 0, 0);
  run_conv(stream, b1, 32, 32, 256, wrb[2], dr1b2, b3, b2, nullptr,
           32, 32, 256, 256, 1, 1, dy11, dx11, 0, 0,
           32, 32, 1, 0, 1, 0, 0, 0, 0, 0, 0, 0, 0);
  run_conv(stream, b2, 32, 32, 256, wra[3], dr2b1, nullptr, b1, nullptr,
           32, 32, 256, 256, 1, 9, dy33, dx33, 1, 1,
           32, 32, 1, 0, 1, 0, 0, 0, 0, 0, 0, 0, 0);
  run_conv(stream, b1, 32, 32, 256, wrb[3], dr2b2, b2, b3, nullptr,
           32, 32, 256, 256, 1, 1, dy11, dx11, 0, 0,
           32, 32, 1, 0, 1, 0, 0, 0, 0, 0, 0, 0, 0);

  // ---- transposed conv 1 (256->256, 4x4 s2 p1, relu) : 4 parity passes ----
  for (int par = 0; par < 4; ++par) {
    int po = par >> 1, pw = par & 1;
    int dyT[4], dxT[4];
    for (int t = 0; t < 4; ++t) {
      int tdy = t >> 1, tdx = t & 1;
      dyT[t] = (po == 0) ? (tdy ? -1 : 0) : (tdy ? 0 : 1);
      dxT[t] = (pw == 0) ? (tdx ? -1 : 0) : (tdx ? 0 : 1);
    }
    run_conv(stream, b3, 32, 32, 256,
             wt1 + (size_t)par * 256 * 4 * 256, dt1b, nullptr, bufA, nullptr,
             32, 32, 256, 256, 1, 4, dyT, dxT, 0, 1,
             64, 64, 2, po, 2, pw, 0, 0, 0, 0, 0, 0, 0);
  }

  // ---- transposed conv 2 (256->3, sigmoid) -> fp32 NCHW x_tilde ----
  for (int par = 0; par < 4; ++par) {
    int po = par >> 1, pw = par & 1;
    int dyT[4], dxT[4];
    for (int t = 0; t < 4; ++t) {
      int tdy = t >> 1, tdx = t & 1;
      dyT[t] = (po == 0) ? (tdy ? -1 : 0) : (tdy ? 0 : 1);
      dxT[t] = (pw == 0) ? (tdx ? -1 : 0) : (tdx ? 0 : 1);
    }
    run_conv(stream, bufA, 64, 64, 256,
             wt2 + (size_t)par * 16 * 4 * 256, dt2b, nullptr, nullptr, out_x,
             64, 64, 3, 16, 1, 4, dyT, dxT, 0, 2,
             128, 128, 2, po, 2, pw, 3, 128, 128, 2, po, 2, pw);
  }
}
